// ModelBase_37220186587581
// MI455X (gfx1250) — compile-verified, matrix operand order checked by reading, not run on hardware
//
#include <hip/hip_runtime.h>
#include <hip/hip_bf16.h>
#include <stdint.h>

// ---------------------------------------------------------------------------
// BiLSTM + span heads, MI455X (gfx1250, wave32, WMMA bf16).
// Pipeline: embed-gather -> weight transpose/bf16 -> Xi = x@Wi + b (WMMA GEMM)
//   -> persistent LSTM recurrence (Wh staged to LDS via TDM tensor_load_to_lds,
//      grid sync per step, WMMA from LDS)
//   -> span pooling -> per-batch obj bias
//   -> fused o-GEMM + log-softmax CE (w_oT staged via global_load_async_to_lds)
//   -> fused s1/s2 dot + BCE -> deterministic final reduction.
// ---------------------------------------------------------------------------

typedef __attribute__((ext_vector_type(16))) __bf16 v16bf;
typedef __attribute__((ext_vector_type(8)))  __bf16 v8bf;
typedef __attribute__((ext_vector_type(8)))  float  v8f;
typedef unsigned u32x4 __attribute__((ext_vector_type(4)));
typedef unsigned u32x8 __attribute__((ext_vector_type(8)));

#define B_   64
#define S_   512
#define H_   512
#define EMB_ 300
#define KIN  320          // EMB padded to multiple of 32
#define G4   2048         // 4*H
#define NCP  64           // NCLS=50 padded to 64
#define BS_  (B_*S_)

// ---------------------------- device helpers ------------------------------

__device__ __forceinline__ v8f wmma_bf(v16bf a, v16bf b, v8f c) {
  // (neg_a, A, neg_b, B, c_mod, C, reuse_a, reuse_b)
  return __builtin_amdgcn_wmma_f32_16x16x32_bf16(false, a, false, b, (short)0, c,
                                                 false, false);
}

// 16-bit A/B fragment: lane holds K-chunks [k0..k0+7] and [k0+16..k0+23].
__device__ __forceinline__ v16bf ldfrag(const __bf16* p) {
  v8bf lo = *(const v8bf*)p;
  v8bf hi = *(const v8bf*)(p + 16);
  return __builtin_shufflevector(lo, hi, 0,1,2,3,4,5,6,7,8,9,10,11,12,13,14,15);
}

__device__ __forceinline__ float sigm(float x)  { return 1.f / (1.f + __expf(-x)); }
__device__ __forceinline__ float tanha(float x) { return 1.f - 2.f / (__expf(2.f*x) + 1.f); }

// generic->LDS byte offset (low 32 bits of the shared aperture address)
__device__ __forceinline__ unsigned ldsoff32(const void* p) {
  return (unsigned)(uintptr_t)p;
}

// One TDM 2D-tile load: tile_d0 x tile_d1 elements of 2 bytes, row pitch
// stride_elems, global src -> contiguous LDS at ldsaddr. D# per ISA 8.3/8.4.
__device__ __forceinline__ void tdm_load_2d_b16(const void* gsrc, unsigned ldsaddr,
                                                unsigned tile_d0, unsigned tile_d1,
                                                unsigned stride_elems) {
  unsigned long long ga = (unsigned long long)(uintptr_t)gsrc;
  u32x4 g0;
  g0[0] = 0x1u;                                      // count=1, user mode
  g0[1] = ldsaddr;                                   // lds_addr
  g0[2] = (unsigned)(ga & 0xFFFFFFFFu);              // global_addr[31:0]
  g0[3] = (unsigned)((ga >> 32) & 0x01FFFFFFu)       // global_addr[56:32]
          | 0x80000000u;                             // type=2 ("image")
  u32x8 g1;
  g1[0] = 0x00010000u;                               // wg_mask=0, data_size=2B
  g1[1] = (tile_d0 & 0xFFFFu) << 16;                 // tensor_dim0[15:0]
  g1[2] = (0x7FFFu) << 16;                           // tensor_dim1[15:0] (ample)
  g1[3] = (tile_d0 & 0xFFFFu) << 16;                 // tile_dim0
  g1[4] = tile_d1 & 0xFFFFu;                         // tile_dim1 (tile_dim2=0)
  g1[5] = stride_elems;                              // tensor_dim0_stride[31:0]
  g1[6] = 0u;
  g1[7] = 0u;
  asm volatile("tensor_load_to_lds %0, %1" :: "s"(g0), "s"(g1) : "memory");
}

// grid-level split barrier over 16 workgroups of one direction group
__device__ __forceinline__ void gsync(unsigned* cnt, unsigned target) {
  __threadfence();
  __syncthreads();
  if (threadIdx.x == 0) {
    __hip_atomic_fetch_add(cnt, 1u, __ATOMIC_ACQ_REL, __HIP_MEMORY_SCOPE_AGENT);
    while (__hip_atomic_load(cnt, __ATOMIC_ACQUIRE, __HIP_MEMORY_SCOPE_AGENT) < target)
      __builtin_amdgcn_s_sleep(2);
  }
  __syncthreads();
}

// ------------------------------- kernels ----------------------------------

__global__ __launch_bounds__(64,1) void k_init(unsigned* sync) {
  if (threadIdx.x < 2) sync[threadIdx.x] = 0u;
}

// maskf[s*B + b] = (text[b,s] != 0)
__global__ __launch_bounds__(256,1) void k_mask(const int* __restrict__ text,
                                               float* __restrict__ maskf) {
  int idx = blockIdx.x * 256 + threadIdx.x;          // s*B+b
  int b = idx & (B_-1), s = idx >> 6;
  maskf[idx] = (text[b*S_ + s] != 0) ? 1.f : 0.f;
}

// xbf[(s*B+b)*KIN + e] = bf16(emb[text[b,s], e]) (zero-padded e>=300)
__global__ __launch_bounds__(256,1) void k_embed(const int* __restrict__ text,
                                                const float* __restrict__ emb,
                                                __bf16* __restrict__ xbf) {
  int idx = blockIdx.x * 256 + threadIdx.x;
  int r = idx / KIN, e = idx % KIN;
  int b = r & (B_-1), s = r >> 6;
  int tok = text[b*S_ + s];
  float v = (e < EMB_) ? emb[tok*EMB_ + e] : 0.f;
  xbf[idx] = (__bf16)v;
}

// dst[n*dstK + k] = bf16(src[k*srcN + n]) with zero padding
__global__ __launch_bounds__(256,1) void k_convT(__bf16* __restrict__ dst,
                                                const float* __restrict__ src,
                                                int srcK, int srcN, int dstK) {
  int idx = blockIdx.x * 256 + threadIdx.x;
  int n = idx / dstK, k = idx % dstK;
  float v = (k < srcK && n < srcN) ? src[k*srcN + n] : 0.f;
  dst[idx] = (__bf16)v;
}

// Xi = xbf(BS x 320) @ WiT'(320 x 2048) + bias, bf16 out, time-major rows.
// One wave = 64x64 output tile, 16 C frags, K loop of 10.
__global__ __launch_bounds__(128,1) void k_gemm_xi(const __bf16* __restrict__ x,
                                                  const __bf16* __restrict__ wT,
                                                  const float* __restrict__ bias,
                                                  __bf16* __restrict__ Xi) {
  int wv = threadIdx.x >> 5, lane = threadIdx.x & 31;
  int gw = blockIdx.x * 4 + wv;                 // 0..16383
  int mblk = gw >> 5, nblk = gw & 31;
  int l16 = lane & 15, hh = (lane < 16) ? 0 : 8;
  const __bf16* arow[4]; const __bf16* brow[4];
#pragma unroll
  for (int i = 0; i < 4; i++) {
    arow[i] = x  + (mblk*64 + i*16 + l16) * KIN;
    brow[i] = wT + (nblk*64 + i*16 + l16) * KIN;
  }
  v8f C[4][4] = {};
#pragma unroll
  for (int kk = 0; kk < 10; kk++) {
    int k0 = kk*32 + hh;
    v16bf a[4], b[4];
#pragma unroll
    for (int i = 0; i < 4; i++) {
      if (kk < 9) __builtin_prefetch(arow[i] + k0 + 32, 0, 3);  // global_prefetch_b8
      a[i] = ldfrag(arow[i] + k0);
      b[i] = ldfrag(brow[i] + k0);
    }
#pragma unroll
    for (int m = 0; m < 4; m++)
#pragma unroll
      for (int n = 0; n < 4; n++) C[m][n] = wmma_bf(a[m], b[n], C[m][n]);
  }
#pragma unroll
  for (int m = 0; m < 4; m++)
#pragma unroll
    for (int n = 0; n < 4; n++)
#pragma unroll
      for (int v = 0; v < 8; v++) {
        int row = mblk*64 + m*16 + v + hh;
        int col = nblk*64 + n*16 + l16;
        Xi[row*G4 + col] = (__bf16)(C[m][n][v] + bias[col]);
      }
}

// Persistent bidirectional LSTM recurrence.
// 32 WGs x 128 thr. blocks 0-15 = forward, 16-31 = backward.
// Wave owns (hidden slice j of 16 cols, batch half of 32). WhT strips are
// DMA'd once into LDS by the Tensor Data Mover; c,h cell state lives in
// VGPRs; h double-buffered in global; grid sync per step.
__global__ __launch_bounds__(128,1) void k_lstm(const __bf16* __restrict__ XiF,
                                               const __bf16* __restrict__ XiB,
                                               const __bf16* __restrict__ WhTF,
                                               const __bf16* __restrict__ WhTB,
                                               const float* __restrict__ maskf,
                                               __bf16* __restrict__ tv,
                                               __bf16* hF, __bf16* hB,
                                               unsigned* sync) {
  extern __shared__ char smem_raw[];                  // 2 slices x 64 KB
  __bf16* lds = (__bf16*)smem_raw;
  int wg  = blockIdx.x;
  int dir = wg >> 4;
  int wgd = wg & 15;
  int wv = threadIdx.x >> 5, lane = threadIdx.x & 31;
  int sl = wv >> 1, mhalf = wv & 1;
  int j = wgd*2 + sl;                                 // hidden slice 0..31
  int l16 = lane & 15, hh = (lane < 16) ? 0 : 8;
  const __bf16* Xi  = dir ? XiB  : XiF;
  const __bf16* WhT = dir ? WhTB : WhTF;
  __bf16* hbase = dir ? hB : hF;                      // 2 phases x 64x512
  unsigned* cnt = sync + dir;
  __bf16* ldsS = lds + sl*32768;                      // row-major gate strips

  // TDM-stage this WG's WhT strips: 2 slices x 4 gates, each 16 rows x K=512.
  if (wv == 0) {
#pragma unroll
    for (int sl2 = 0; sl2 < 2; sl2++) {
      int jj = wgd*2 + sl2;
#pragma unroll
      for (int g = 0; g < 4; g++) {
        const __bf16* gsrc = WhT + (size_t)(g*H_ + jj*16) * H_;
        unsigned laddr = ldsoff32(lds + sl2*32768 + g*8192);
        tdm_load_2d_b16(gsrc, laddr, /*tile_d0=*/H_, /*tile_d1=*/16,
                        /*stride=*/H_);
      }
    }
    __builtin_amdgcn_s_wait_tensorcnt(0);
  }
  // zero h phase-0 (each WG zeros its 1/16 of the 64x512 buffer)
  {
    v8bf z = {};
    __bf16* d = hbase + wgd*2048 + threadIdx.x*16;
    *(v8bf*)d = z; *(v8bf*)(d + 8) = z;
  }
  gsync(cnt, 16u);

  float cst[2][8] = {}, hst[2][8] = {};

  for (int t = 0; t < S_; t++) {
    int spos = dir ? (S_-1-t) : t;
    const __bf16* hrd = hbase + (t & 1) * (B_*H_);
    __bf16*       hwr = hbase + ((t & 1) ^ 1) * (B_*H_);
    v8f C[2][4] = {};
#pragma unroll
    for (int kk = 0; kk < 16; kk++) {
      int k0 = kk*32 + hh;
      v16bf a[2], b[4];
#pragma unroll
      for (int m = 0; m < 2; m++) {
        int row = mhalf*32 + m*16 + l16;
        a[m] = ldfrag(hrd + row*H_ + k0);
      }
#pragma unroll
      for (int g = 0; g < 4; g++)
        b[g] = ldfrag(ldsS + g*8192 + l16*512 + k0);   // ds_load from TDM tile
#pragma unroll
      for (int m = 0; m < 2; m++)
#pragma unroll
        for (int g = 0; g < 4; g++) C[m][g] = wmma_bf(a[m], b[g], C[m][g]);
    }
    // gate epilogue: z = hWh + Xi(t); cell update, masked, store h slice
#pragma unroll
    for (int m = 0; m < 2; m++) {
#pragma unroll
      for (int v = 0; v < 8; v++) {
        int bb = mhalf*32 + m*16 + v + hh;
        int xr = (spos*B_ + bb)*G4 + j*16 + l16;
        float zi = C[m][0][v] + (float)Xi[xr];
        float zf = C[m][1][v] + (float)Xi[xr + 512];
        float zg = C[m][2][v] + (float)Xi[xr + 1024];
        float zo = C[m][3][v] + (float)Xi[xr + 1536];
        float iv = sigm(zi), fv = sigm(zf), gv = tanha(zg), ov = sigm(zo);
        float cn = fv*cst[m][v] + iv*gv;
        float hn = ov*tanha(cn);
        float mk = maskf[spos*B_ + bb];
        cst[m][v] += mk * (cn - cst[m][v]);
        hst[m][v] += mk * (hn - hst[m][v]);
        __bf16 hv = (__bf16)hst[m][v];
        hwr[bb*H_ + j*16 + l16] = hv;
        tv[(spos*B_ + bb)*1024 + dir*H_ + j*16 + l16] = hv;
      }
    }
    gsync(cnt, 16u * (unsigned)(t + 2));
  }
}

// sbj_vec[b] = mean over s in [lo,hi] of tv[s,b,:]
__global__ __launch_bounds__(256,1) void k_pool(const __bf16* __restrict__ tv,
                                               const int* __restrict__ bound,
                                               float* __restrict__ sbj) {
  int b = blockIdx.x;
  int lo = bound[b*2], hi = bound[b*2 + 1];
  float inv = 1.f / (float)(hi - lo + 1);
  for (int col = threadIdx.x; col < 1024; col += 256) {
    float acc = 0.f;
    for (int s = lo; s <= hi; s++) acc += (float)tv[(s*B_ + b)*1024 + col];
    sbj[b*1024 + col] = acc * inv;
  }
}

// objb[b,n] = sbj_vec[b] . w_o[1024:2048, n] + b_o[n]   (pad cols -> -1e30)
__global__ __launch_bounds__(64,1) void k_objbias(const float* __restrict__ sbj,
                                                 const float* __restrict__ wo,
                                                 const float* __restrict__ bo,
                                                 float* __restrict__ ob) {
  int b = blockIdx.x, n = threadIdx.x;
  float acc = -1.0e30f;
  if (n < 50) {
    acc = bo[n];
    const float* sv = sbj + b*1024;
    for (int k = 0; k < 1024; k++) acc += sv[k] * wo[(1024 + k)*50 + n];
  }
  ob[b*NCP + n] = acc;
}

// Fused o-GEMM (M=16/wave, N=64, K=1024 WMMA) + log-softmax CE, partial/wave.
// w_oT (128 KB) is staged once per WG into LDS with async-to-LDS loads; the
// first 16 KB of the block is reused as the epilogue scratch after a barrier.
__global__ __launch_bounds__(128,1) void k_ogemm_ce(const __bf16* __restrict__ tv,
                                                   const __bf16* __restrict__ woT,
                                                   const float* __restrict__ objb,
                                                   const int* __restrict__ labels,
                                                   const float* __restrict__ maskf,
                                                   float* __restrict__ pCE) {
  extern __shared__ char smem_raw[];                 // 128 KB
  __bf16* lbw = (__bf16*)smem_raw;
  int wv = threadIdx.x >> 5, lane = threadIdx.x & 31;
  int gw = blockIdx.x * 4 + wv;                  // 0..2047
  int rowbase = gw * 16;
  int l16 = lane & 15, hh = (lane < 16) ? 0 : 8;

  // async-stage woT: 65536 bf16 = 128 threads x 64 iters x b128
  {
    unsigned lbase = ldsoff32(lbw);
    for (int it = 0; it < 64; it++) {
      int e = (it*128 + threadIdx.x) * 8;          // element index
      unsigned laddr = lbase + (unsigned)e * 2u;
      unsigned long long ga = (unsigned long long)(uintptr_t)(woT + e);
      asm volatile("global_load_async_to_lds_b128 %0, %1, off"
                   :: "v"(laddr), "v"(ga) : "memory");
    }
    asm volatile("s_wait_asynccnt 0x0" ::: "memory");
  }
  __syncthreads();

  const __bf16* arow = tv + (rowbase + l16) * 1024;
  v8f C[4] = {};
#pragma unroll
  for (int kk = 0; kk < 32; kk++) {
    int k0 = kk*32 + hh;
    v16bf a = ldfrag(arow + k0);
#pragma unroll
    for (int n = 0; n < 4; n++) {
      v16bf b = ldfrag(lbw + (n*16 + l16)*1024 + k0);  // ds_load from LDS
      C[n] = wmma_bf(a, b, C[n]);
    }
  }
  __syncthreads();                                  // all waves done with B
  float* sm = (float*)smem_raw;                     // reuse 16 KB for epilogue
#pragma unroll
  for (int n = 0; n < 4; n++)
#pragma unroll
    for (int v = 0; v < 8; v++)
      sm[wv*1024 + (v + hh)*64 + n*16 + l16] = C[n][v];
  __syncthreads();

  int row = lane & 15, half = lane >> 4;
  int rg = rowbase + row;
  int b = rg & (B_-1), s = rg >> 6;
  float mk = maskf[rg];
  int label = labels[b*S_ + s];
  const float* zr = sm + wv*1024 + row*64;
  const float* ob = objb + b*NCP;
  float mx = -3.0e38f;
  for (int c = half*32; c < half*32 + 32; c++) mx = fmaxf(mx, zr[c] + ob[c]);
  mx = fmaxf(mx, __shfl_xor(mx, 16, 32));
  float se = 0.f, lv = 0.f;
  for (int c = half*32; c < half*32 + 32; c++) {
    float z = zr[c] + ob[c];
    se += __expf(z - mx);
    lv += (c == label) ? z : 0.f;
  }
  se += __shfl_xor(se, 16, 32);
  lv += __shfl_xor(lv, 16, 32);
  float ce = (half == 0) ? (mk * (-(lv - mx - __logf(se)))) : 0.f;
#pragma unroll
  for (int d = 1; d < 32; d <<= 1) ce += __shfl_xor(ce, d, 32);
  if (lane == 0) pCE[gw] = ce;
}

// Fused s1/s2 dot (K=1024) + BCE; one wave per (b,s) row; partial per row.
__global__ __launch_bounds__(256,1) void k_sbce(const __bf16* __restrict__ tv,
                                               const float* __restrict__ ws1,
                                               const float* __restrict__ bs1,
                                               const float* __restrict__ ws2,
                                               const float* __restrict__ bs2,
                                               const int* __restrict__ ystart,
                                               const int* __restrict__ yend,
                                               const float* __restrict__ maskf,
                                               float* __restrict__ pBCE,
                                               float* __restrict__ pCNT) {
  int wv = threadIdx.x >> 5, lane = threadIdx.x & 31;
  int row = blockIdx.x * 8 + wv;
  float d1 = 0.f, d2 = 0.f;
  const __bf16* tr = tv + row * 1024;
  for (int i = 0; i < 32; i++) {
    int k = lane + i*32;
    float t = (float)tr[k];
    d1 += t * ws1[k];
    d2 += t * ws2[k];
  }
#pragma unroll
  for (int d = 1; d < 32; d <<= 1) { d1 += __shfl_xor(d1, d, 32); d2 += __shfl_xor(d2, d, 32); }
  if (lane == 0) {
    int b = row & (B_-1), s = row >> 6;
    float mk = maskf[row];
    float x1 = d1 + bs1[0], x2 = d2 + bs2[0];
    float y1 = (float)ystart[b*S_ + s], y2 = (float)yend[b*S_ + s];
    // log(sigmoid(x)) = -log1p(exp(-x)), clipped at -100 like the reference
    float lp1  = fmaxf(-log1pf(__expf(-x1)), -100.f);
    float l1m1 = fmaxf(-log1pf(__expf( x1)), -100.f);
    float lp2  = fmaxf(-log1pf(__expf(-x2)), -100.f);
    float l1m2 = fmaxf(-log1pf(__expf( x2)), -100.f);
    float bce1 = -(y1*lp1 + (1.f - y1)*l1m1);
    float bce2 = -(y2*lp2 + (1.f - y2)*l1m2);
    pBCE[row] = mk * (bce1 + bce2);
    pCNT[row] = mk;
  }
}

__device__ __forceinline__ float blkred(float v, float* sm) {
  int tid = threadIdx.x;
  sm[tid] = v; __syncthreads();
  for (int s = 128; s > 0; s >>= 1) {
    if (tid < s) sm[tid] += sm[tid + s];
    __syncthreads();
  }
  float r = sm[0]; __syncthreads();
  return r;
}

__global__ __launch_bounds__(256,1) void k_final(const float* __restrict__ pBCE,
                                                const float* __restrict__ pCNT,
                                                const float* __restrict__ pCE1,
                                                const float* __restrict__ pCE2,
                                                float* __restrict__ out) {
  __shared__ float sm[256];
  int tid = threadIdx.x;
  float a = 0.f, cnt = 0.f, c1 = 0.f, c2 = 0.f;
  for (int i = tid; i < BS_; i += 256) { a += pBCE[i]; cnt += pCNT[i]; }
  for (int i = tid; i < 2048; i += 256) { c1 += pCE1[i]; c2 += pCE2[i]; }
  a   = blkred(a, sm);
  cnt = blkred(cnt, sm);
  c1  = blkred(c1, sm);
  c2  = blkred(c2, sm);
  if (tid == 0) out[0] = (2.f*a + c1 + c2) / cnt;
}

// ------------------------------- host side --------------------------------

extern "C" void kernel_launch(void* const* d_in, const int* in_sizes, int n_in,
                              void* d_out, int out_size, void* d_ws, size_t ws_size,
                              hipStream_t stream) {
  (void)in_sizes; (void)n_in; (void)out_size; (void)ws_size;
  const int*   text      = (const int*)d_in[0];
  const int*   sbj_start = (const int*)d_in[1];
  const int*   sbj_end   = (const int*)d_in[2];
  const int*   sbj_bound = (const int*)d_in[3];
  const int*   obj_start = (const int*)d_in[4];
  const int*   obj_end   = (const int*)d_in[5];
  const float* emb  = (const float*)d_in[6];
  const float* Wi_f = (const float*)d_in[7];
  const float* Wh_f = (const float*)d_in[8];
  const float* b_f  = (const float*)d_in[9];
  const float* Wi_b = (const float*)d_in[10];
  const float* Wh_b = (const float*)d_in[11];
  const float* b_b  = (const float*)d_in[12];
  const float* w_s1 = (const float*)d_in[13];
  const float* b_s1 = (const float*)d_in[14];
  const float* w_s2 = (const float*)d_in[15];
  const float* b_s2 = (const float*)d_in[16];
  const float* w_o1 = (const float*)d_in[17];
  const float* b_o1 = (const float*)d_in[18];
  const float* w_o2 = (const float*)d_in[19];
  const float* b_o2 = (const float*)d_in[20];

  char* ws = (char*)d_ws;
  size_t off = 0;
  auto take = [&](size_t bytes) -> void* {
    void* p = ws + off;
    off += (bytes + 255) & ~(size_t)255;
    return p;
  };
  __bf16* xbf   = (__bf16*)take((size_t)BS_ * KIN * 2);
  __bf16* WiTF  = (__bf16*)take((size_t)G4 * KIN * 2);
  __bf16* WiTB  = (__bf16*)take((size_t)G4 * KIN * 2);
  __bf16* WhTF  = (__bf16*)take((size_t)G4 * H_ * 2);
  __bf16* WhTB  = (__bf16*)take((size_t)G4 * H_ * 2);
  __bf16* XiF   = (__bf16*)take((size_t)BS_ * G4 * 2);
  __bf16* XiB   = (__bf16*)take((size_t)BS_ * G4 * 2);
  __bf16* tv    = (__bf16*)take((size_t)BS_ * 1024 * 2);
  __bf16* hF    = (__bf16*)take((size_t)2 * B_ * H_ * 2);   // double buffered
  __bf16* hB    = (__bf16*)take((size_t)2 * B_ * H_ * 2);
  float*  maskf = (float*)take((size_t)BS_ * 4);
  float*  sbj   = (float*)take((size_t)B_ * 1024 * 4);
  float*  objb1 = (float*)take((size_t)B_ * NCP * 4);
  float*  objb2 = (float*)take((size_t)B_ * NCP * 4);
  __bf16* woT1  = (__bf16*)take((size_t)NCP * 1024 * 2);
  __bf16* woT2  = (__bf16*)take((size_t)NCP * 1024 * 2);
  float*  pBCE  = (float*)take((size_t)BS_ * 4);
  float*  pCNT  = (float*)take((size_t)BS_ * 4);
  float*  pCE1  = (float*)take((size_t)2048 * 4);
  float*  pCE2  = (float*)take((size_t)2048 * 4);
  unsigned* syncc = (unsigned*)take(256);

  k_init<<<1, 64, 0, stream>>>(syncc);
  k_mask<<<BS_/256, 256, 0, stream>>>(text, maskf);
  k_embed<<<(BS_*KIN)/256, 256, 0, stream>>>(text, emb, xbf);

  k_convT<<<(G4*KIN)/256, 256, 0, stream>>>(WiTF, Wi_f, EMB_, G4, KIN);
  k_convT<<<(G4*KIN)/256, 256, 0, stream>>>(WiTB, Wi_b, EMB_, G4, KIN);
  k_convT<<<(G4*H_)/256, 256, 0, stream>>>(WhTF, Wh_f, H_, G4, H_);
  k_convT<<<(G4*H_)/256, 256, 0, stream>>>(WhTB, Wh_b, H_, G4, H_);
  k_convT<<<(NCP*1024)/256, 256, 0, stream>>>(woT1, w_o1, 1024, 50, 1024);
  k_convT<<<(NCP*1024)/256, 256, 0, stream>>>(woT2, w_o2, 1024, 50, 1024);

  // Xi = x @ Wi + b, both directions: 16384 waves of 64x64 tiles
  k_gemm_xi<<<4096, 128, 0, stream>>>(xbf, WiTF, b_f, XiF);
  k_gemm_xi<<<4096, 128, 0, stream>>>(xbf, WiTB, b_b, XiB);

  // persistent recurrence: 32 WGs x 128 threads, 128 KB dynamic LDS each
  k_lstm<<<32, 128, 131072, stream>>>(XiF, XiB, WhTF, WhTB, maskf, tv, hF, hB, syncc);

  k_pool<<<B_, 256, 0, stream>>>(tv, sbj_bound, sbj);
  k_objbias<<<B_, 64, 0, stream>>>(sbj, w_o1, b_o1, objb1);
  k_objbias<<<B_, 64, 0, stream>>>(sbj, w_o2, b_o2, objb2);

  k_sbce<<<BS_/8, 256, 0, stream>>>(tv, w_s1, b_s1, w_s2, b_s2,
                                    sbj_start, sbj_end, maskf, pBCE, pCNT);

  k_ogemm_ce<<<512, 128, 131072, stream>>>(tv, woT1, objb1, obj_start, maskf, pCE1);
  k_ogemm_ce<<<512, 128, 131072, stream>>>(tv, woT2, objb2, obj_end,   maskf, pCE2);

  k_final<<<1, 256, 0, stream>>>(pBCE, pCNT, pCE1, pCE2, (float*)d_out);
}
